// VariationalLSTM_69836168233438
// MI455X (gfx1250) — matrix operand order repaired and checked by reading, not compiled, not run on hardware
//
#include <hip/hip_runtime.h>
#include <hip/hip_bf16.h>

// Problem dimensions (fixed by the reference)
#define B_ 128
#define T_ 256
#define I_ 256
#define H_ 1024
#define G_ 4096 // 4*H

typedef __attribute__((ext_vector_type(16))) __bf16 v16bf;
typedef __attribute__((ext_vector_type(8)))  float  v8f;

// ---------------------------------------------------------------------------
// WMMA fragment loaders (bf16, 16x16x32). Layouts per CDNA5 ISA 7.12.2:
//  A (16x32, MxK): lane l -> row M=l%16; K chunk {koff..koff+7} in V0..3 and
//                  {koff+16..koff+23} in V4..7, koff = (l/16)*8.
//  B (32x16, KxN): lane l -> col N=l%16; same K chunks as A (koff..+7, koff+16..+23).
// Both become two 16-byte global loads per lane from row-major bf16 storage.
// ---------------------------------------------------------------------------
__device__ __forceinline__ v16bf load_frag_a(const __hip_bfloat16* __restrict__ A,
                                             size_t lda, int row0, int k0, int lane) {
    const __hip_bfloat16* p =
        A + (size_t)(row0 + (lane & 15)) * lda + (size_t)(k0 + ((lane >> 4) << 3));
    union { v16bf v; uint4 q[2]; } u;
    u.q[0] = *(const uint4*)(p);        // K: k0+koff   .. +7
    u.q[1] = *(const uint4*)(p + 16);   // K: k0+koff+16.. +23
    return u.v;
}

__device__ __forceinline__ v16bf load_frag_b(const __hip_bfloat16* __restrict__ W,
                                             size_t ldw, int col0, int k0, int lane) {
    const __hip_bfloat16* p =
        W + (size_t)(col0 + (lane & 15)) * ldw + (size_t)(k0 + ((lane >> 4) << 3));
    union { v16bf v; uint4 q[2]; } u;
    u.q[0] = *(const uint4*)(p);        // K: k0+koff   .. +7
    u.q[1] = *(const uint4*)(p + 16);   // K: k0+koff+16.. +23
    return u.v;
}

__device__ __forceinline__ v8f wmma_bf16(v16bf a, v16bf b, v8f c) {
    // (neg_a, A, neg_b, B, c_mod, C, reuse_a, reuse_b)
    return __builtin_amdgcn_wmma_f32_16x16x32_bf16(false, a, false, b, (short)0, c,
                                                   false, false);
}

// ---------------------------------------------------------------------------
// fp32 -> bf16 pack
// ---------------------------------------------------------------------------
__global__ void k_pack_bf16(const float* __restrict__ in,
                            __hip_bfloat16* __restrict__ out, long n) {
    long i = (long)blockIdx.x * blockDim.x + threadIdx.x;
    if (i < n) out[i] = __float2bfloat16(in[i]);
}

// ---------------------------------------------------------------------------
// Batched input projection: C[M,G] = A[M,K](bf16) @ W[G,K]^T(bf16) + b1 + b2
// 8 waves/block, each wave owns a 32x64 fp32 tile. 8 WMMAs per K-step.
// ---------------------------------------------------------------------------
__global__ __launch_bounds__(256) void k_gemm_wmma(
    const __hip_bfloat16* __restrict__ A, int lda,
    const __hip_bfloat16* __restrict__ W, int ldw, int K,
    const float* __restrict__ b1, const float* __restrict__ b2,
    float* __restrict__ C, int ldc) {
    const int lane = threadIdx.x & 31;
    const int wave = threadIdx.x >> 5;
    const int n0 = blockIdx.x * 64;
    const int m0 = (blockIdx.y * 8 + wave) * 32;

    v8f acc[2][4] = {};
    for (int k0 = 0; k0 < K; k0 += 32) {
        v16bf a0 = load_frag_a(A, (size_t)lda, m0,      k0, lane);
        v16bf a1 = load_frag_a(A, (size_t)lda, m0 + 16, k0, lane);
#pragma unroll
        for (int nj = 0; nj < 4; ++nj) {
            v16bf bf = load_frag_b(W, (size_t)ldw, n0 + nj * 16, k0, lane);
            acc[0][nj] = wmma_bf16(a0, bf, acc[0][nj]);
            acc[1][nj] = wmma_bf16(a1, bf, acc[1][nj]);
        }
    }
    // C/D layout: lane l, VGPR r -> M = r + 8*(l/16), N = l%16
    const int mo  = (lane >> 4) << 3;
    const int nlo = lane & 15;
#pragma unroll
    for (int mi = 0; mi < 2; ++mi)
#pragma unroll
        for (int nj = 0; nj < 4; ++nj)
#pragma unroll
            for (int r = 0; r < 8; ++r) {
                int m = m0 + mi * 16 + r + mo;
                int n = n0 + nj * 16 + nlo;
                C[(size_t)m * ldc + n] = acc[mi][nj][r] + b1[n] + b2[n];
            }
}

// ---------------------------------------------------------------------------
// One LSTM timestep, fused: gates = xg_t + h_prev @ W_hh^T, then cell update.
// Block = 4 waves, one per gate; each wave -> 32(batch) x 64(h) WMMA tile over
// K=1024. Gate tiles meet in LDS; elementwise epilogue fused after barrier.
// Grid: (H/64, B/32) = (16, 4).
// ---------------------------------------------------------------------------
__global__ __launch_bounds__(128) void k_lstm_step(
    const __hip_bfloat16* __restrict__ hprev, size_t ldh,
    const __hip_bfloat16* __restrict__ W,          // [4H,H] bf16, ldw = H
    const float* __restrict__ xg, size_t xg_stride, // base already at timestep t
    float* __restrict__ cbuf,                       // [B,H] fp32 cell state
    __hip_bfloat16* __restrict__ hout, size_t ldhout,
    float* __restrict__ hout_f32, size_t ldhf,      // nullable
    float* __restrict__ hfin, float* __restrict__ cfin, int write_final) {
    __shared__ float lds[32 * 4 * 64]; // [m_local][gate][h_local], 32 KB

    const int lane = threadIdx.x & 31;
    const int gate = threadIdx.x >> 5;
    const int hb = blockIdx.x * 64;
    const int mb = blockIdx.y * 32;
    const int n0 = gate * H_ + hb;

    v8f acc[2][4] = {};
    for (int k0 = 0; k0 < H_; k0 += 32) {
        v16bf a0 = load_frag_a(hprev, ldh, mb,      k0, lane);
        v16bf a1 = load_frag_a(hprev, ldh, mb + 16, k0, lane);
#pragma unroll
        for (int nj = 0; nj < 4; ++nj) {
            v16bf bf = load_frag_b(W, (size_t)H_, n0 + nj * 16, k0, lane);
            acc[0][nj] = wmma_bf16(a0, bf, acc[0][nj]);
            acc[1][nj] = wmma_bf16(a1, bf, acc[1][nj]);
        }
    }
    const int mo  = (lane >> 4) << 3;
    const int nlo = lane & 15;
#pragma unroll
    for (int mi = 0; mi < 2; ++mi)
#pragma unroll
        for (int nj = 0; nj < 4; ++nj)
#pragma unroll
            for (int r = 0; r < 8; ++r) {
                int m  = mi * 16 + r + mo;   // local batch row 0..31
                int hc = nj * 16 + nlo;      // local h col    0..63
                lds[(m * 4 + gate) * 64 + hc] = acc[mi][nj][r];
            }
    __syncthreads();

    for (int idx = threadIdx.x; idx < 32 * 64; idx += 128) {
        int m  = idx >> 6;
        int hc = idx & 63;
        int b  = mb + m;
        int h  = hb + hc;
        const float* xr = xg + (size_t)b * xg_stride + h;
        float vi = lds[(m * 4 + 0) * 64 + hc] + xr[0 * H_];
        float vf = lds[(m * 4 + 1) * 64 + hc] + xr[1 * H_];
        float vg = lds[(m * 4 + 2) * 64 + hc] + xr[2 * H_];
        float vo = lds[(m * 4 + 3) * 64 + hc] + xr[3 * H_];
        float ig = 1.0f / (1.0f + __expf(-vi));
        float fg = 1.0f / (1.0f + __expf(-vf));
        float gg = tanhf(vg);
        float og = 1.0f / (1.0f + __expf(-vo));
        size_t cix = (size_t)b * H_ + h;
        float cn = fg * cbuf[cix] + ig * gg;
        cbuf[cix] = cn;
        float hn = og * tanhf(cn);
        hout[(size_t)b * ldhout + h] = __float2bfloat16(hn);
        if (hout_f32) hout_f32[(size_t)b * ldhf + h] = hn;
        if (write_final) { hfin[cix] = hn; cfin[cix] = cn; }
    }
}

// ---------------------------------------------------------------------------
extern "C" void kernel_launch(void* const* d_in, const int* in_sizes, int n_in,
                              void* d_out, int out_size, void* d_ws, size_t ws_size,
                              hipStream_t stream) {
    const float* x     = (const float*)d_in[0];
    const float* hx0   = (const float*)d_in[1];
    const float* cx0   = (const float*)d_in[2];
    const float* W_ih0 = (const float*)d_in[3];
    const float* W_hh0 = (const float*)d_in[4];
    const float* b_ih0 = (const float*)d_in[5];
    const float* b_hh0 = (const float*)d_in[6];
    const float* W_ih1 = (const float*)d_in[7];
    const float* W_hh1 = (const float*)d_in[8];
    const float* b_ih1 = (const float*)d_in[9];
    const float* b_hh1 = (const float*)d_in[10];
    (void)in_sizes; (void)n_in; (void)out_size; (void)ws_size;

    // Workspace carve (256B-aligned bump allocator)
    uint8_t* p = (uint8_t*)d_ws;
    auto carve = [&p](size_t bytes) -> void* {
        void* r = (void*)p;
        p += (bytes + 255) & ~(size_t)255;
        return r;
    };
    float*          xg   = (float*)         carve((size_t)B_ * T_ * G_ * 4); // 512 MB
    __hip_bfloat16* Wpi  = (__hip_bfloat16*)carve((size_t)G_ * H_ * 2);      // packed W_ih
    __hip_bfloat16* Wph  = (__hip_bfloat16*)carve((size_t)G_ * H_ * 2);      // packed W_hh
    __hip_bfloat16* xb   = (__hip_bfloat16*)carve((size_t)B_ * T_ * I_ * 2); // packed x
    __hip_bfloat16* ysb  = (__hip_bfloat16*)carve((size_t)B_ * T_ * H_ * 2); // bf16 h stream
    __hip_bfloat16* h0b  = (__hip_bfloat16*)carve((size_t)B_ * H_ * 2);      // packed h0
    float*          cbuf = (float*)         carve((size_t)B_ * H_ * 4);      // cell state

    float* out_x = (float*)d_out;
    float* out_h = out_x + (size_t)B_ * T_ * H_;
    float* out_c = out_h + (size_t)B_ * H_;

    auto pack = [&](const float* src, __hip_bfloat16* dst, long n) {
        k_pack_bf16<<<(unsigned)((n + 255) / 256), 256, 0, stream>>>(src, dst, n);
    };

    const dim3 gemm_grid(G_ / 64, (B_ * T_ / 32) / 8);
    const dim3 step_grid(H_ / 64, B_ / 32);

    // ---------------- Layer 0 ----------------
    pack(x, xb, (long)B_ * T_ * I_);
    pack(W_ih0, Wpi, (long)G_ * I_);
    pack(W_hh0, Wph, (long)G_ * H_);
    k_gemm_wmma<<<gemm_grid, 256, 0, stream>>>(xb, I_, Wpi, I_, I_, b_ih0, b_hh0, xg, G_);
    pack(hx0, h0b, (long)B_ * H_);
    hipMemcpyAsync(cbuf, cx0, (size_t)B_ * H_ * 4, hipMemcpyDeviceToDevice, stream);
    for (int t = 0; t < T_; ++t) {
        const __hip_bfloat16* hp = t ? (ysb + (size_t)(t - 1) * H_) : h0b;
        size_t ldh = t ? (size_t)T_ * H_ : (size_t)H_;
        k_lstm_step<<<step_grid, 128, 0, stream>>>(
            hp, ldh, Wph, xg + (size_t)t * G_, (size_t)T_ * G_, cbuf,
            ysb + (size_t)t * H_, (size_t)T_ * H_,
            (float*)nullptr, (size_t)0, (float*)nullptr, (float*)nullptr, 0);
    }

    // ---------------- Layer 1 ----------------
    pack(W_ih1, Wpi, (long)G_ * H_);
    pack(W_hh1, Wph, (long)G_ * H_);
    // Projection consumes ysb (layer-0 outputs, bf16) fully before the scan
    // below starts overwriting it with layer-1 h's (same stream => ordered).
    k_gemm_wmma<<<gemm_grid, 256, 0, stream>>>(ysb, H_, Wpi, H_, H_, b_ih1, b_hh1, xg, G_);
    pack(hx0 + (size_t)B_ * H_, h0b, (long)B_ * H_);
    hipMemcpyAsync(cbuf, cx0 + (size_t)B_ * H_, (size_t)B_ * H_ * 4,
                   hipMemcpyDeviceToDevice, stream);
    for (int t = 0; t < T_; ++t) {
        const __hip_bfloat16* hp = t ? (ysb + (size_t)(t - 1) * H_) : h0b;
        size_t ldh = t ? (size_t)T_ * H_ : (size_t)H_;
        k_lstm_step<<<step_grid, 128, 0, stream>>>(
            hp, ldh, Wph, xg + (size_t)t * G_, (size_t)T_ * G_, cbuf,
            ysb + (size_t)t * H_, (size_t)T_ * H_,
            out_x + (size_t)t * H_, (size_t)T_ * H_, out_h, out_c,
            (t == T_ - 1) ? 1 : 0);
    }
}
